// Position_Attention_7610682049053
// MI455X (gfx1250) — matrix operand order repaired and checked by reading, not compiled, not run on hardware
//
#include <hip/hip_runtime.h>

typedef __attribute__((ext_vector_type(16))) _Float16 v16h;
typedef __attribute__((ext_vector_type(8)))  _Float16 v8h;
typedef __attribute__((ext_vector_type(8)))  float    v8f;
typedef __attribute__((ext_vector_type(4)))  float    v4f;

#define BB  16
#define NN  2048
#define CC  512
#define DD  64
#define QKD 128   // Q (64) and K (64) packed per token row

// ---- WMMA fragment loaders (16x16x32 f16; this code's contiguous k = 16*(lane/16) + s for A and B) ----

// A-matrix 16x32 (MxK), source row-major with leading dim `lda` (halves).
// lane<16: row=lane,   halves 0..15 -> K=0..15
// lane>=16: row=lane-16, halves 0..15 -> K=16..31
__device__ __forceinline__ v16h load_a16(const _Float16* base, int lda) {
  const int lane = threadIdx.x & 31;
  const int row  = lane & 15;
  const int k0   = (lane < 16) ? 0 : 16;
  const _Float16* p = base + row * lda + k0;
  v8h lo = *(const v8h*)(p);
  v8h hi = *(const v8h*)(p + 8);
  v16h r;
#pragma unroll
  for (int i = 0; i < 8; ++i) { r[i] = lo[i]; r[8 + i] = hi[i]; }
  return r;
}

// B-matrix 32x16 (KxN) loaded from a source stored as [N rows, K cols] row-major
// (i.e. B = source^T). lane<16: col=lane, halves -> K=0..15;
// lane>=16: col=lane-16, halves -> K=16..31. One contiguous 32B load per lane.
__device__ __forceinline__ v16h load_b16_t(const _Float16* base, int ld) {
  const int lane = threadIdx.x & 31;
  const int col  = lane & 15;
  const int k0   = (lane < 16) ? 0 : 16;
  return *(const v16h*)(base + col * ld + k0);
}

__device__ __forceinline__ v8f wmma_f16(v16h a, v16h b, v8f c) {
  return __builtin_amdgcn_wmma_f32_16x16x32_f16(false, a, false, b,
                                                (short)0, c, false, false);
}

// ---- Kernel 1: f32 -> f16 conversion (n divisible by 4) ----
__global__ void __launch_bounds__(256)
cvt_kernel(const float* __restrict__ src, _Float16* __restrict__ dst, int n) {
  int i = (blockIdx.x * 256 + threadIdx.x) * 4;
  if (i + 3 < n) {
    v4f v = *(const v4f*)(src + i);
    dst[i + 0] = (_Float16)v[0];
    dst[i + 1] = (_Float16)v[1];
    dst[i + 2] = (_Float16)v[2];
    dst[i + 3] = (_Float16)v[3];
  }
}

// ---- Kernel 2: QK projection.  QK[b*N+n][d] = x[n,:] @ Wqk[d,:]^T + bias
// grid.x = B*N/16 row tiles; wave w (0..7) = output col tile (128 cols total)
__global__ void __launch_bounds__(256)
qk_proj_kernel(const _Float16* __restrict__ xh, const _Float16* __restrict__ Wqk,
               const float* __restrict__ bq, const float* __restrict__ bk,
               _Float16* __restrict__ QK) {
  const int rt   = blockIdx.x;
  const int w    = threadIdx.x >> 5;
  const int lane = threadIdx.x & 31;
  const _Float16* abase = xh  + (size_t)rt * 16 * CC;
  const _Float16* bbase = Wqk + (size_t)w  * 16 * CC;
  v8f acc = {};
#pragma unroll 4
  for (int k = 0; k < CC; k += 32) {
    v16h a = load_a16(abase + k, CC);
    v16h b = load_b16_t(bbase + k, CC);   // Wqk rows are output cols
    acc = wmma_f16(a, b, acc);
  }
  const int col  = lane & 15;
  const int row0 = (lane < 16) ? 0 : 8;
  const int dd   = w * 16 + col;
  const float bias = (dd < DD) ? bq[dd] : bk[dd - DD];
  _Float16* out = QK + (size_t)rt * 16 * QKD + dd;
#pragma unroll
  for (int g = 0; g < 8; ++g)
    out[(size_t)(row0 + g) * QKD] = (_Float16)(acc[g] + bias);
}

// ---- Kernel 3: V projection, output transposed: Vt[b][c][n] = Wv[c,:]·x[n,:] + bv[c]
// grid = (N/128, C/16, B); wave w handles n-tile blockIdx.x*8+w
__global__ void __launch_bounds__(256)
v_proj_kernel(const _Float16* __restrict__ xh, const _Float16* __restrict__ Wvh,
              const float* __restrict__ bv, _Float16* __restrict__ Vt) {
  const int b    = blockIdx.z;
  const int ct   = blockIdx.y;
  const int w    = threadIdx.x >> 5;
  const int nt   = blockIdx.x * 8 + w;
  const int lane = threadIdx.x & 31;
  const _Float16* abase = Wvh + (size_t)ct * 16 * CC;
  const _Float16* bbase = xh + ((size_t)b * NN + (size_t)nt * 16) * CC;
  v8f acc = {};
#pragma unroll 4
  for (int k = 0; k < CC; k += 32) {
    v16h a = load_a16(abase + k, CC);       // rows = output channels c
    v16h bf = load_b16_t(bbase + k, CC);    // x rows = output cols n
    acc = wmma_f16(a, bf, acc);
  }
  const int col  = lane & 15;
  const int row0 = (lane < 16) ? 0 : 8;
  const int n = nt * 16 + col;
#pragma unroll
  for (int g = 0; g < 8; ++g) {
    const int c = ct * 16 + row0 + g;
    Vt[((size_t)b * CC + c) * NN + n] = (_Float16)(acc[g] + bv[c]);
  }
}

// ---- Kernel 4: fused attention.  One block = one 16-query strip of one batch.
// Phase 1: S[16 x 2048] = Q Kt (wave w -> keys [w*256, w*256+256))
// Phase 2: in-LDS softmax per row, fully vectorized (b128 LDS traffic only)
// Phase 3: out[16 x 512] = P Vt^T, wave w -> channels [w*64, w*64+64); fused residual
__global__ void __launch_bounds__(256)
attn_kernel(const _Float16* __restrict__ QK, const _Float16* __restrict__ Vt,
            const float* __restrict__ x, const float* __restrict__ gamma_p,
            float* __restrict__ out) {
  __shared__ __align__(32) _Float16 S[16 * NN];   // 64 KB score/prob strip

  const int qt   = blockIdx.x;
  const int b    = blockIdx.y;
  const int w    = threadIdx.x >> 5;
  const int lane = threadIdx.x & 31;
  const int col  = lane & 15;
  const int row0 = (lane < 16) ? 0 : 8;

  // Q fragments for this 16-row strip (K dims 0..63); dead after phase 1
  {
    const _Float16* qbase = QK + ((size_t)b * NN + (size_t)qt * 16) * QKD;
    const v16h qa0 = load_a16(qbase + 0,  QKD);
    const v16h qa1 = load_a16(qbase + 32, QKD);

    // -------- Phase 1: energies --------
#pragma unroll 2
    for (int mt = 0; mt < 16; ++mt) {
      const int mbase = w * 256 + mt * 16;
      const _Float16* kbase = QK + ((size_t)b * NN + mbase) * QKD + DD;
      v16h kb0 = load_b16_t(kbase + 0,  QKD);   // key rows -> B columns
      v16h kb1 = load_b16_t(kbase + 32, QKD);
      v8f s = {};
      s = wmma_f16(qa0, kb0, s);
      s = wmma_f16(qa1, kb1, s);
#pragma unroll
      for (int g = 0; g < 8; ++g)
        S[(row0 + g) * NN + mbase + col] = (_Float16)s[g];
    }
  }
  __syncthreads();

  // -------- Phase 2: softmax; 16 threads per row, each owns 128 contiguous
  // halves held in registers (16 x v8h): one b128 read + one b128 write of LDS.
  {
    const int row = threadIdx.x >> 4;
    const int sub = threadIdx.x & 15;
    v8h* rp = (v8h*)(S + row * NN + sub * 128);
    v8h vv[16];
#pragma unroll
    for (int t = 0; t < 16; ++t) vv[t] = rp[t];

    float m = -3.0e38f;
#pragma unroll
    for (int t = 0; t < 16; ++t)
#pragma unroll
      for (int j = 0; j < 8; ++j) m = fmaxf(m, (float)vv[t][j]);
#pragma unroll
    for (int o = 1; o < 16; o <<= 1) m = fmaxf(m, __shfl_xor(m, o, 16));

    float sum = 0.f;
#pragma unroll
    for (int t = 0; t < 16; ++t)
#pragma unroll
      for (int j = 0; j < 8; ++j) {
        float e = __expf((float)vv[t][j] - m);
        sum += e;
        vv[t][j] = (_Float16)e;
      }
#pragma unroll
    for (int o = 1; o < 16; o <<= 1) sum += __shfl_xor(sum, o, 16);
    const float inv = 1.0f / sum;

#pragma unroll
    for (int t = 0; t < 16; ++t) {
      v8h e = vv[t];
#pragma unroll
      for (int j = 0; j < 8; ++j) e[j] = (_Float16)((float)e[j] * inv);
      rp[t] = e;
    }
  }
  __syncthreads();

  // -------- Phase 3: P @ V^T + fused gamma/residual epilogue --------
  const int cw = w * 64;
  v8f acc[4] = {};
#pragma unroll 2
  for (int km = 0; km < NN; km += 32) {
    v16h pa = load_a16(S + km, NN);           // prob fragment from LDS
#pragma unroll
    for (int ct = 0; ct < 4; ++ct) {
      const _Float16* vb = Vt + ((size_t)b * CC + cw + ct * 16) * NN + km;
      v16h bf = load_b16_t(vb, NN);           // Vt rows (channels) -> B columns
      acc[ct] = wmma_f16(pa, bf, acc[ct]);
    }
  }
  const float gamma = gamma_p[0];
#pragma unroll
  for (int ct = 0; ct < 4; ++ct) {
#pragma unroll
    for (int g = 0; g < 8; ++g) {
      const int n = qt * 16 + row0 + g;
      const int c = cw + ct * 16 + col;
      const size_t idx = ((size_t)b * NN + n) * CC + c;
      out[idx] = gamma * acc[ct][g] + x[idx];
    }
  }
}

// ---- host launcher ----
extern "C" void kernel_launch(void* const* d_in, const int* in_sizes, int n_in,
                              void* d_out, int out_size, void* d_ws, size_t ws_size,
                              hipStream_t stream) {
  const float* x  = (const float*)d_in[0];
  const float* Wq = (const float*)d_in[1];
  const float* bq = (const float*)d_in[2];
  const float* Wk = (const float*)d_in[3];
  const float* bk = (const float*)d_in[4];
  const float* Wv = (const float*)d_in[5];
  const float* bv = (const float*)d_in[6];
  const float* gm = (const float*)d_in[7];
  float* out = (float*)d_out;

  char* ws = (char*)d_ws;
  size_t off = 0;
  auto carve = [&](size_t bytes) -> char* {
    char* p = ws + off;
    off = (off + bytes + 255) & ~(size_t)255;
    return p;
  };
  _Float16* xh  = (_Float16*)carve((size_t)BB * NN * CC * 2);   // 32 MB
  _Float16* Wqk = (_Float16*)carve((size_t)QKD * CC * 2);       // 128 KB
  _Float16* Wvh = (_Float16*)carve((size_t)CC * CC * 2);        // 512 KB
  _Float16* QK  = (_Float16*)carve((size_t)BB * NN * QKD * 2);  // 8 MB
  _Float16* Vt  = (_Float16*)carve((size_t)BB * CC * NN * 2);   // 32 MB
  (void)ws_size; (void)in_sizes; (void)n_in; (void)out_size;

  const int nx = BB * NN * CC;
  cvt_kernel<<<(nx / 4 + 255) / 256, 256, 0, stream>>>(x, xh, nx);
  cvt_kernel<<<(DD * CC / 4 + 255) / 256, 256, 0, stream>>>(Wq, Wqk, DD * CC);
  cvt_kernel<<<(DD * CC / 4 + 255) / 256, 256, 0, stream>>>(Wk, Wqk + DD * CC, DD * CC);
  cvt_kernel<<<(CC * CC / 4 + 255) / 256, 256, 0, stream>>>(Wv, Wvh, CC * CC);

  qk_proj_kernel<<<BB * NN / 16, 256, 0, stream>>>(xh, Wqk, bq, bk, QK);
  v_proj_kernel<<<dim3(NN / 128, CC / 16, BB), 256, 0, stream>>>(xh, Wvh, bv, Vt);
  attn_kernel<<<dim3(NN / 16, BB), 256, 0, stream>>>(QK, Vt, x, gm, out);
}
